// BaseAttention_55044300865964
// MI455X (gfx1250) — matrix operand order repaired and checked by reading, not compiled, not run on hardware
//
#include <hip/hip_runtime.h>
#include <hip/hip_bf16.h>
#include <math.h>

// ---- CDNA5 (gfx1250) flash attention, bf16 WMMA with fp32 accumulation ----
// B=4, H=16, S=2048, D=64, fp32 in/out. 64-key kv tiles, software-pipelined
// staging, softmax denominator computed on the matrix pipe (P x ones).

typedef __attribute__((ext_vector_type(16))) __bf16 v16bf;
typedef __attribute__((ext_vector_type(8)))  float  v8f;
typedef __attribute__((ext_vector_type(4)))  float  f32x4;
typedef __attribute__((ext_vector_type(4)))  unsigned int u32x4;

union FragAB {            // 16 bf16 per lane == one WMMA A or B operand (8 VGPRs)
    v16bf v;
    u32x4 u[2];
};

constexpr int   Bc = 4, Hc = 16, Sc = 2048, Dc = 64;
constexpr float SCALE = 0.125f;       // 1/sqrt(64)
constexpr float NEGV  = -10000.0f;

// padded LDS strides (bf16 elements); all fragment reads stay 16B-aligned
constexpr int QSTR = 72;   // 144 B/row
constexpr int KSTR = 72;
constexpr int VSTR = 72;   // V stored transposed: [d][k], 64 k-cols + pad
constexpr int PSTR = 72;   // P: 16 rows x 64 cols + pad, per wave

__global__ __launch_bounds__(128)
void fa_fwd_gfx1250(const float* __restrict__ qg_,
                    const float* __restrict__ kg_,
                    const float* __restrict__ vg_,
                    const int*   __restrict__ maskg,
                    float*       __restrict__ outg)
{
    __shared__ __align__(16) __bf16 sQ[64 * QSTR];       // 9216 B
    __shared__ __align__(16) __bf16 sK[64 * KSTR];       // 9216 B
    __shared__ __align__(16) __bf16 sVt[64 * VSTR];      // 9216 B
    __shared__ __align__(16) __bf16 sP[4][16 * PSTR];    // 9216 B

    const int tid  = threadIdx.x;
    const int wv   = tid >> 5;          // wave id 0..3 (wave32)
    const int lane = tid & 31;
    const int n    = lane & 15;         // N column / row-within-tile index
    const bool hi  = lane >= 16;
    const int hi16 = hi ? 16 : 0;       // unused; B loads use bs8, same K order as A
    const int bs8  = hi ? 8  : 0;       // A-layout K offset / C-layout M offset

    const int bid = blockIdx.x;
    const int qt  = bid & 31;           // 32 query tiles of 64 rows per (b,h)
    const int bh  = bid >> 5;           // b*H + h
    const int b   = bh / Hc;

    const float* qg = qg_ + ((size_t)bh * Sc + (size_t)qt * 64) * Dc;
    float*       og = outg + ((size_t)bh * Sc + (size_t)qt * 64) * Dc;
    const int*   mg = maskg + (size_t)b * Sc;
    const float* kbase = kg_ + (size_t)bh * Sc * Dc;
    const float* vbase = vg_ + (size_t)bh * Sc * Dc;

    // ---- stage Q tile (64x64 fp32 -> bf16) into LDS, cooperatively ----
    for (int i = tid; i < (64 * 64) / 4; i += 128) {
        const int e = i * 4, r = e >> 6, c = e & 63;
        f32x4 f = *(const f32x4*)(qg + r * Dc + c);
        __bf16* d = &sQ[r * QSTR + c];
        d[0] = (__bf16)f.x; d[1] = (__bf16)f.y; d[2] = (__bf16)f.z; d[3] = (__bf16)f.w;
    }
    __syncthreads();

    // ---- per-wave Q A-fragments (row = lane%16, bf16 16x32 A layout) ----
    const int qrow = wv * 16 + n;
    FragAB aq0, aq1;
    aq0.u[0] = *(const u32x4*)&sQ[qrow * QSTR +  0 + bs8];   // K 0-7  / 8-15
    aq0.u[1] = *(const u32x4*)&sQ[qrow * QSTR + 16 + bs8];   // K 16-23 / 24-31
    aq1.u[0] = *(const u32x4*)&sQ[qrow * QSTR + 32 + bs8];   // K 32-39 / 40-47
    aq1.u[1] = *(const u32x4*)&sQ[qrow * QSTR + 48 + bs8];   // K 48-55 / 56-63

    // all-ones B matrix (bf16 1.0 = 0x3F80) for rowsum-by-WMMA
    FragAB bone;
    const unsigned int one2 = 0x3F803F80u;
    bone.u[0] = u32x4{one2, one2, one2, one2};
    bone.u[1] = u32x4{one2, one2, one2, one2};

    v8f acc[4] = {v8f{}, v8f{}, v8f{}, v8f{}};
    v8f l_acc  = {};
    float m_r[8];
    #pragma unroll
    for (int r = 0; r < 8; ++r) m_r[r] = -INFINITY;

    // ---- software-pipelined staging: prologue loads tile 0 into registers ----
    f32x4 kreg[8], vreg[8];
    #pragma unroll
    for (int i = 0; i < 8; ++i) {
        const int e = (tid + i * 128) * 4;             // 0..4095 over 64x64 tile
        kreg[i] = *(const f32x4*)(kbase + e);
        vreg[i] = *(const f32x4*)(vbase + e);
    }

    for (int kb = 0; kb < Sc; kb += 64) {
        // ---- commit staged registers (fp32 -> bf16) to LDS ----
        #pragma unroll
        for (int i = 0; i < 8; ++i) {
            const int e = (tid + i * 128) * 4, r = e >> 6, c = e & 63;
            __bf16* dk = &sK[r * KSTR + c];
            dk[0] = (__bf16)kreg[i].x; dk[1] = (__bf16)kreg[i].y;
            dk[2] = (__bf16)kreg[i].z; dk[3] = (__bf16)kreg[i].w;
            sVt[(c + 0) * VSTR + r] = (__bf16)vreg[i].x;
            sVt[(c + 1) * VSTR + r] = (__bf16)vreg[i].y;
            sVt[(c + 2) * VSTR + r] = (__bf16)vreg[i].z;
            sVt[(c + 3) * VSTR + r] = (__bf16)vreg[i].w;
        }
        __syncthreads();

        // ---- issue next tile's global loads; latency hidden behind WMMAs ----
        if (kb + 64 < Sc) {
            const float* kg = kbase + (size_t)(kb + 64) * Dc;
            const float* vg = vbase + (size_t)(kb + 64) * Dc;
            #pragma unroll
            for (int i = 0; i < 8; ++i) {
                const int e = (tid + i * 128) * 4;
                kreg[i] = *(const f32x4*)(kg + e);
                vreg[i] = *(const f32x4*)(vg + e);
            }
            if (kb + 128 < Sc)
                __builtin_prefetch(kbase + (size_t)(kb + 128) * Dc + tid * 16, 0, 1);
        }

        // ---- additive mask bias per 16-key score tile column ----
        float bias[4];
        #pragma unroll
        for (int j = 0; j < 4; ++j)
            bias[j] = (float)mg[kb + 16 * j + n] * NEGV;

        // ---- S = Q K^T : four 16x16 score tiles over 64 keys ----
        v8f sc[4];
        #pragma unroll
        for (int j = 0; j < 4; ++j) {
            FragAB bk;
            v8f s = {};
            bk.u[0] = *(const u32x4*)&sK[(16 * j + n) * KSTR +  0 + bs8];
            bk.u[1] = *(const u32x4*)&sK[(16 * j + n) * KSTR + 16 + bs8];
            s = __builtin_amdgcn_wmma_f32_16x16x32_bf16(false, aq0.v, false, bk.v,
                                                        (short)0, s, false, false);
            bk.u[0] = *(const u32x4*)&sK[(16 * j + n) * KSTR + 32 + bs8];
            bk.u[1] = *(const u32x4*)&sK[(16 * j + n) * KSTR + 48 + bs8];
            s = __builtin_amdgcn_wmma_f32_16x16x32_bf16(false, aq1.v, false, bk.v,
                                                        (short)0, s, false, false);
            sc[j] = s;
        }

        // ---- online softmax: row max via half-wave ds_bpermute reduction ----
        #pragma unroll
        for (int r = 0; r < 8; ++r) {
            float x0 = sc[0][r] * SCALE + bias[0];
            float x1 = sc[1][r] * SCALE + bias[1];
            float x2 = sc[2][r] * SCALE + bias[2];
            float x3 = sc[3][r] * SCALE + bias[3];
            float c = fmaxf(fmaxf(x0, x1), fmaxf(x2, x3));
            c = fmaxf(c, __shfl_xor(c, 1));
            c = fmaxf(c, __shfl_xor(c, 2));
            c = fmaxf(c, __shfl_xor(c, 4));
            c = fmaxf(c, __shfl_xor(c, 8));
            const float mn    = fmaxf(m_r[r], c);
            const float alpha = __expf(m_r[r] - mn);
            m_r[r] = mn;
            const float p0 = __expf(x0 - mn);
            const float p1 = __expf(x1 - mn);
            const float p2 = __expf(x2 - mn);
            const float p3 = __expf(x3 - mn);
            // rescale running accumulators (sum handled on the matrix pipe)
            l_acc[r]  *= alpha;
            acc[0][r] *= alpha; acc[1][r] *= alpha;
            acc[2][r] *= alpha; acc[3][r] *= alpha;
            // spill P (C layout: row M = r + bs8, col N = 16j + n)
            const int rowM = r + bs8;
            sP[wv][rowM * PSTR +  0 + n] = (__bf16)p0;
            sP[wv][rowM * PSTR + 16 + n] = (__bf16)p1;
            sP[wv][rowM * PSTR + 32 + n] = (__bf16)p2;
            sP[wv][rowM * PSTR + 48 + n] = (__bf16)p3;
        }
        // intra-wave LDS RAW: P stores above -> A-fragment loads below
        asm volatile("s_wait_dscnt 0" ::: "memory");

        // P as two 16x32 A-fragments (K = keys 0-31 and 32-63 of this tile)
        FragAB ap0, ap1;
        ap0.u[0] = *(const u32x4*)&sP[wv][n * PSTR +  0 + bs8];
        ap0.u[1] = *(const u32x4*)&sP[wv][n * PSTR + 16 + bs8];
        ap1.u[0] = *(const u32x4*)&sP[wv][n * PSTR + 32 + bs8];
        ap1.u[1] = *(const u32x4*)&sP[wv][n * PSTR + 48 + bs8];

        // l += rowsum(P) on the matrix pipe: P x ones
        l_acc = __builtin_amdgcn_wmma_f32_16x16x32_bf16(false, ap0.v, false, bone.v,
                                                        (short)0, l_acc, false, false);
        l_acc = __builtin_amdgcn_wmma_f32_16x16x32_bf16(false, ap1.v, false, bone.v,
                                                        (short)0, l_acc, false, false);

        // ---- O += P V : four 16-wide d-chunks, two K=32 steps each ----
        #pragma unroll
        for (int t = 0; t < 4; ++t) {
            FragAB bv;
            bv.u[0] = *(const u32x4*)&sVt[(t * 16 + n) * VSTR +  0 + bs8];
            bv.u[1] = *(const u32x4*)&sVt[(t * 16 + n) * VSTR + 16 + bs8];
            acc[t] = __builtin_amdgcn_wmma_f32_16x16x32_bf16(false, ap0.v, false, bv.v,
                                                             (short)0, acc[t], false, false);
            bv.u[0] = *(const u32x4*)&sVt[(t * 16 + n) * VSTR + 32 + bs8];
            bv.u[1] = *(const u32x4*)&sVt[(t * 16 + n) * VSTR + 48 + bs8];
            acc[t] = __builtin_amdgcn_wmma_f32_16x16x32_bf16(false, ap1.v, false, bv.v,
                                                             (short)0, acc[t], false, false);
        }
        __syncthreads();   // protect sK/sVt before next iteration's staging
    }

    // ---- normalize and store (C/D layout: row M = r + bs8, col N = n) ----
    #pragma unroll
    for (int r = 0; r < 8; ++r) {
        const float inv = 1.0f / l_acc[r];
        const int row = wv * 16 + r + bs8;
        #pragma unroll
        for (int t = 0; t < 4; ++t)
            og[row * Dc + t * 16 + n] = acc[t][r] * inv;
    }
}

extern "C" void kernel_launch(void* const* d_in, const int* in_sizes, int n_in,
                              void* d_out, int out_size, void* d_ws, size_t ws_size,
                              hipStream_t stream) {
    (void)in_sizes; (void)n_in; (void)out_size; (void)d_ws; (void)ws_size;
    const float* q    = (const float*)d_in[0];
    const float* k    = (const float*)d_in[1];
    const float* v    = (const float*)d_in[2];
    const int*   mask = (const int*)d_in[3];
    float*       out  = (float*)d_out;

    const int blocks = Bc * Hc * (Sc / 64);   // 2048 blocks, 64 query rows each
    fa_fwd_gfx1250<<<blocks, 128, 0, stream>>>(q, k, v, mask, out);
}
